// CNN_ternary_28552942584105
// MI455X (gfx1250) — 8-bit matrix operand order repaired and checked by reading, compiled, not run on hardware
//
#include <hip/hip_runtime.h>
#include <cstdint>
#include <cstddef>


#define DELTA_ 0.1f
#define EPS_   1e-5f

typedef int          v16i  __attribute__((ext_vector_type(16)));
typedef float        v8f   __attribute__((ext_vector_type(8)));
typedef unsigned int u32x4 __attribute__((ext_vector_type(4)));
typedef int          i32x4 __attribute__((ext_vector_type(4)));
typedef int          i32x8 __attribute__((ext_vector_type(8)));

#if defined(__has_builtin)
#  if __has_builtin(__builtin_amdgcn_tensor_load_to_lds) && \
      __has_builtin(__builtin_amdgcn_s_wait_tensorcnt)
#    define USE_TDM 1
#  endif
#endif
#ifndef USE_TDM
#  define USE_TDM 0
#endif

__device__ __forceinline__ uint8_t tern_fp8(float v, float d) {
    return (v >= d) ? (uint8_t)0x38 : ((v <= -d) ? (uint8_t)0xB8 : (uint8_t)0x00);
}
__device__ __forceinline__ float tern_f(float v, float d) {
    return (v >= d) ? 1.0f : ((v <= -d) ? -1.0f : 0.0f);
}
__device__ __forceinline__ float htanh(float v) {
    return fminf(fmaxf(v, -1.0f), 1.0f);
}

#if USE_TDM
__device__ __forceinline__ void tdm_load_b_tile(uint64_t gaddr, int N, int K) {
    u32x4 g0;
    g0[0] = 1u;
    g0[1] = 0u;
    g0[2] = (uint32_t)gaddr;
    g0[3] = (uint32_t)(gaddr >> 32) | (2u << 30);
    i32x8 g1;
    g1[0] = 0;
    g1[1] = (int)(((unsigned)K & 0xffffu) << 16);
    g1[2] = (int)(((unsigned)K >> 16) |
                  (((unsigned)N & 0xffffu) << 16));
    g1[3] = (int)(128u << 16);
    g1[4] = N;
    g1[5] = K;
    g1[6] = 0;
    g1[7] = 0;
    i32x4 z4 = {0, 0, 0, 0};
#if __clang_major__ >= 23
    i32x8 z8 = {0, 0, 0, 0, 0, 0, 0, 0};
    __builtin_amdgcn_tensor_load_to_lds(g0, g1, z4, z4, z8, 0);
#else
    __builtin_amdgcn_tensor_load_to_lds(g0, g1, z4, z4, 0);
#endif
}
#endif

__global__ void reduce_max_k(const float* __restrict__ w, int n, float* __restrict__ out) {
    __shared__ float sh[256];
    float m = -3.402823e38f;
    for (int i = threadIdx.x; i < n; i += 256) m = fmaxf(m, w[i]);
    sh[threadIdx.x] = m;
    __syncthreads();
    for (int st = 128; st > 0; st >>= 1) {
        if (threadIdx.x < st) sh[threadIdx.x] = fmaxf(sh[threadIdx.x], sh[threadIdx.x + st]);
        __syncthreads();
    }
    if (threadIdx.x == 0) out[0] = DELTA_ * sh[0];
}

__global__ void tern_w_fp8_k(const float* __restrict__ w, const float* __restrict__ dptr,
                             uint8_t* __restrict__ out, int N, int K, int Kpad) {
    int idx = blockIdx.x * 256 + threadIdx.x;
    if (idx >= N * Kpad) return;
    int n = idx / Kpad, k = idx - n * Kpad;
    uint8_t v = 0;
    if (k < K) v = tern_fp8(w[n * K + k], dptr[0]);
    out[idx] = v;
}

__global__ void tern_f32_k(const float* __restrict__ s, const float* __restrict__ dptr,
                           float* __restrict__ out, int n) {
    int i = blockIdx.x * 256 + threadIdx.x;
    if (i < n) out[i] = tern_f(s[i], dptr[0]);
}

__global__ void conv1_k(const float* __restrict__ x, const float* __restrict__ tw1,
                        const float* __restrict__ tb1, float* __restrict__ h1) {
    int idx = blockIdx.x * 256 + threadIdx.x;
    int wo = idx & 63;
    int t = idx >> 6;
    int h = t % 6; t /= 6;
    int c = t & 31; int n = t >> 5;
    const float* xr = x + ((size_t)n * 6 + h) * 128;
    const float* wr = tw1 + c * 9;
    float acc = 0.f;
    int wbase = wo * 2 - 4;
#pragma unroll
    for (int kw = 0; kw < 9; ++kw) {
        int wi = wbase + kw;
        if (wi >= 0 && wi < 128) acc += xr[wi] * wr[kw];
    }
    h1[idx] = acc + tb1[c];
}

__global__ void stats_init_k(float* sums, float* sqs) {
    sums[threadIdx.x] = 0.f;
    sqs[threadIdx.x] = 0.f;
}

__global__ void stats_k(const float* __restrict__ y, int Nn, int C, int S,
                        float* __restrict__ sums, float* __restrict__ sqs) {
    int c = blockIdx.y;
    long long total = (long long)Nn * S;
    float s1 = 0.f, s2 = 0.f;
    for (long long i = (long long)blockIdx.x * 256 + threadIdx.x; i < total;
         i += (long long)gridDim.x * 256) {
        long long n = i / S;
        long long s = i - n * S;
        float v = y[((n * C + c) * S) + s];
        s1 += v; s2 += v * v;
    }
    __shared__ float sh1[256], sh2[256];
    int t = threadIdx.x;
    sh1[t] = s1; sh2[t] = s2;
    __syncthreads();
    for (int st = 128; st > 0; st >>= 1) {
        if (t < st) { sh1[t] += sh1[t + st]; sh2[t] += sh2[t + st]; }
        __syncthreads();
    }
    if (t == 0) { atomicAdd(&sums[c], sh1[0]); atomicAdd(&sqs[c], sh2[0]); }
}

__global__ void bn1_pool_tern_k(const float* __restrict__ h1,
                                const float* __restrict__ g, const float* __restrict__ be,
                                const float* __restrict__ sums, const float* __restrict__ sqs,
                                const float* __restrict__ dptr, uint8_t* __restrict__ a2) {
    int idx = blockIdx.x * 256 + threadIdx.x;
    int w = idx & 31;
    int t = idx >> 5;
    int h = t % 6; t /= 6;
    int c = t & 31; int n = t >> 5;
    const float cnt = 4096.f * 6.f * 64.f;
    float mean = sums[c] / cnt;
    float var = sqs[c] / cnt - mean * mean;
    float sc = rsqrtf(var + EPS_) * g[c];
    float bo = be[c] - mean * sc;
    size_t base = ((((size_t)n * 32 + c) * 6 + h) * 64) + 2 * w;
    float v0 = htanh(h1[base] * sc + bo);
    float v1 = htanh(h1[base + 1] * sc + bo);
    a2[idx] = tern_fp8(fmaxf(v0, v1), dptr[0]);
}

__global__ void bn2_tern_k(const float* __restrict__ y2,
                           const float* __restrict__ g, const float* __restrict__ be,
                           const float* __restrict__ sums, const float* __restrict__ sqs,
                           const float* __restrict__ dptr, uint8_t* __restrict__ a3) {
    int idx = blockIdx.x * 256 + threadIdx.x;
    int w = idx & 31;
    int t = idx >> 5;
    int h = t % 6; t /= 6;
    int c = t & 63; int n = t >> 6;
    const float cnt = 786432.f;
    float mean = sums[c] / cnt;
    float var = sqs[c] / cnt - mean * mean;
    float sc = rsqrtf(var + EPS_) * g[c];
    float bo = be[c] - mean * sc;
    float v = y2[((((size_t)n * 6 + h) * 32 + w) * 64) + c];
    a3[idx] = tern_fp8(htanh(v * sc + bo), dptr[0]);
}

__global__ void bn3_pool_tern_k(const float* __restrict__ y3,
                                const float* __restrict__ g, const float* __restrict__ be,
                                const float* __restrict__ sums, const float* __restrict__ sqs,
                                const float* __restrict__ dptr, uint8_t* __restrict__ a4) {
    int idx = blockIdx.x * 256 + threadIdx.x;
    int w = idx & 15;
    int t = idx >> 4;
    int h = t % 6; t /= 6;
    int c = t & 127; int n = t >> 7;
    const float cnt = 786432.f;
    float mean = sums[c] / cnt;
    float var = sqs[c] / cnt - mean * mean;
    float sc = rsqrtf(var + EPS_) * g[c];
    float bo = be[c] - mean * sc;
    size_t rb = ((((size_t)n * 6 + h) * 32 + 2 * w) * 128) + c;
    float v0 = htanh(y3[rb] * sc + bo);
    float v1 = htanh(y3[rb + 128] * sc + bo);
    a4[idx] = tern_fp8(fmaxf(v0, v1), dptr[0]);
}

__global__ void bn4_tern_k(const float* __restrict__ y4,
                           const float* __restrict__ g, const float* __restrict__ be,
                           const float* __restrict__ sums, const float* __restrict__ sqs,
                           const float* __restrict__ dptr, uint8_t* __restrict__ AF) {
    int idx = blockIdx.x * 256 + threadIdx.x;
    int w = idx & 15;
    int t = idx >> 4;
    int c = t & 127; int n = t >> 7;
    const float cnt = 65536.f;
    float mean = sums[c] / cnt;
    float var = sqs[c] / cnt - mean * mean;
    float sc = rsqrtf(var + EPS_) * g[c];
    float bo = be[c] - mean * sc;
    float v = y4[(((size_t)n * 16 + w) * 128) + c];
    AF[(size_t)n * 2048 + c * 16 + w] = tern_fp8(htanh(v * sc + bo), dptr[0]);
}

__global__ void im2col2_k(const uint8_t* __restrict__ a2, uint8_t* __restrict__ A) {
    int idx = blockIdx.x * 256 + threadIdx.x;
    int k = idx & 127;
    int m = idx >> 7;
    uint8_t v = 0;
    if (k < 96) {
        int w = m & 31;
        int t = m >> 5;
        int h = t % 6; int n = t / 6;
        int ci = k / 3, kw = k - ci * 3;
        int wi = w + kw - 1;
        if (wi >= 0 && wi < 32) v = a2[((((size_t)n * 32 + ci) * 6 + h) * 32) + wi];
    }
    A[idx] = v;
}

__global__ void im2col3_k(const uint8_t* __restrict__ a3, uint8_t* __restrict__ A) {
    int idx = blockIdx.x * 256 + threadIdx.x;
    int k = idx & 255;
    int m = idx >> 8;
    uint8_t v = 0;
    if (k < 192) {
        int w = m & 31;
        int t = m >> 5;
        int h = t % 6; int n = t / 6;
        int ci = k / 3, kw = k - ci * 3;
        int wi = w + kw - 1;
        if (wi >= 0 && wi < 32) v = a3[((((size_t)n * 64 + ci) * 6 + h) * 32) + wi];
    }
    A[idx] = v;
}

__global__ void im2col4_k(const uint8_t* __restrict__ a4, uint8_t* __restrict__ A) {
    int idx = blockIdx.x * 256 + threadIdx.x;
    int m = idx / 768;
    int k = idx - m * 768;
    int w = m & 15; int n = m >> 4;
    int ci = k / 6; int kh = k - ci * 6;
    A[idx] = a4[((((size_t)n * 128 + ci) * 6 + kh) * 16) + w];
}

template <int NT>
__global__ __launch_bounds__(256) void gemm_fp8_k(
    const uint8_t* __restrict__ A, const uint8_t* __restrict__ B,
    const float* __restrict__ bias, float* __restrict__ D, int M, int K) {
    constexpr int N = NT * 16;
    __shared__ uint8_t sB[N * 128];

    (void)M;
    int lane = threadIdx.x & 31;
    int wave = threadIdx.x >> 5;
    int m0 = (blockIdx.x * 8 + wave) << 4;
    int row = lane & 15;
    int grp = lane >> 4;

    const uint8_t* Ap = A + (size_t)(m0 + row) * K;

    v8f acc[NT] = {};

    for (int kk = 0; kk < K; kk += 128) {
#if USE_TDM
        if (wave == 0) {
            tdm_load_b_tile((uint64_t)(B + kk), N, K);
            __builtin_amdgcn_s_wait_tensorcnt(0);
        }
#else
        {
            constexpr int CH = (N * 128) / 16;
            for (int c = threadIdx.x; c < CH; c += 256) {
                int col = c >> 3, off = (c & 7) * 16;
                *(uint4*)(sB + col * 128 + off) =
                    *(const uint4*)(B + (size_t)col * K + kk + off);
            }
        }
#endif
        __syncthreads();

        const int* pa = (const int*)(Ap + kk);
        __builtin_prefetch(Ap + kk + 128, 0, 1);
        v16i a;
#pragma unroll
        for (int half = 0; half < 2; ++half) {
            int o = half * 16 + grp * 4;
            a[half * 8 + 0] = pa[o + 0];
            a[half * 8 + 1] = pa[o + 1];
            a[half * 8 + 2] = pa[o + 2];
            a[half * 8 + 3] = pa[o + 3];
            a[half * 8 + 4] = pa[o + 8];
            a[half * 8 + 5] = pa[o + 9];
            a[half * 8 + 6] = pa[o + 10];
            a[half * 8 + 7] = pa[o + 11];
        }

#pragma unroll
        for (int nt = 0; nt < NT; ++nt) {
            const uint8_t* bp = sB + (nt * 16 + row) * 128 + grp * 16;
            v16i b;
#pragma unroll
            for (int q = 0; q < 4; ++q) {
                uint4 t4 = *(const uint4*)(bp + q * 32);
                b[q * 4 + 0] = (int)t4.x;
                b[q * 4 + 1] = (int)t4.y;
                b[q * 4 + 2] = (int)t4.z;
                b[q * 4 + 3] = (int)t4.w;
            }
            acc[nt] = __builtin_amdgcn_wmma_f32_16x16x128_fp8_fp8(
                a, b, (short)0, acc[nt], false, false);
        }
        __syncthreads();
    }

#pragma unroll
    for (int nt = 0; nt < NT; ++nt) {
        float bv = bias[nt * 16 + row];
#pragma unroll
        for (int r = 0; r < 8; ++r) {
            int gm = m0 + r + grp * 8;
            D[(size_t)gm * N + nt * 16 + row] = acc[nt][r] + bv;
        }
    }
}

extern "C" void kernel_launch(void* const* d_in, const int* in_sizes, int n_in,
                              void* d_out, int out_size, void* d_ws, size_t ws_size,
                              hipStream_t stream) {
    (void)in_sizes; (void)n_in; (void)out_size; (void)ws_size;
    const float* x   = (const float*)d_in[0];
    const float* w1  = (const float*)d_in[1];
    const float* b1  = (const float*)d_in[2];
    const float* g1  = (const float*)d_in[3];
    const float* be1 = (const float*)d_in[4];
    const float* w2  = (const float*)d_in[5];
    const float* b2  = (const float*)d_in[6];
    const float* g2  = (const float*)d_in[7];
    const float* be2 = (const float*)d_in[8];
    const float* w3  = (const float*)d_in[9];
    const float* b3  = (const float*)d_in[10];
    const float* g3  = (const float*)d_in[11];
    const float* be3 = (const float*)d_in[12];
    const float* w4  = (const float*)d_in[13];
    const float* b4  = (const float*)d_in[14];
    const float* g4  = (const float*)d_in[15];
    const float* be4 = (const float*)d_in[16];
    const float* wf  = (const float*)d_in[17];
    const float* bf  = (const float*)d_in[18];
    float* out = (float*)d_out;

    char* ws = (char*)d_ws;
    float*   thr  = (float*)(ws + 0);
    float*   sums = (float*)(ws + 256);
    float*   sqs  = (float*)(ws + 1024);
    float*   tb1  = (float*)(ws + 2048);
    float*   tb2  = (float*)(ws + 2304);
    float*   tb3  = (float*)(ws + 2560);
    float*   tb4  = (float*)(ws + 3072);
    float*   tbf  = (float*)(ws + 3584);
    float*   tw1  = (float*)(ws + 4096);
    uint8_t* tw2B = (uint8_t*)(ws + 8192);
    uint8_t* tw3B = (uint8_t*)(ws + 16384);
    uint8_t* tw4B = (uint8_t*)(ws + 49152);
    uint8_t* twfB = (uint8_t*)(ws + 147456);
    float*   bufA = (float*)(ws + (16ull << 20));
    uint8_t* act  = (uint8_t*)(ws + (432ull << 20));
    uint8_t* im   = (uint8_t*)(ws + (496ull << 20));

    reduce_max_k<<<1, 256, 0, stream>>>(w1, 32 * 9, thr + 0);
    reduce_max_k<<<1, 256, 0, stream>>>(w2, 64 * 32 * 3, thr + 1);
    reduce_max_k<<<1, 256, 0, stream>>>(w3, 128 * 64 * 3, thr + 2);
    reduce_max_k<<<1, 256, 0, stream>>>(w4, 128 * 128 * 6, thr + 3);
    reduce_max_k<<<1, 256, 0, stream>>>(wf, 128 * 2048, thr + 4);

    tern_f32_k<<<2, 256, 0, stream>>>(w1, thr + 0, tw1, 288);
    tern_f32_k<<<1, 256, 0, stream>>>(b1, thr + 0, tb1, 32);
    tern_w_fp8_k<<<(64 * 128) / 256, 256, 0, stream>>>(w2, thr + 1, tw2B, 64, 96, 128);
    tern_f32_k<<<1, 256, 0, stream>>>(b2, thr + 1, tb2, 64);
    tern_w_fp8_k<<<(128 * 256) / 256, 256, 0, stream>>>(w3, thr + 2, tw3B, 128, 192, 256);
    tern_f32_k<<<1, 256, 0, stream>>>(b3, thr + 2, tb3, 128);
    tern_w_fp8_k<<<(128 * 768) / 256, 256, 0, stream>>>(w4, thr + 3, tw4B, 128, 768, 768);
    tern_f32_k<<<1, 256, 0, stream>>>(b4, thr + 3, tb4, 128);
    tern_w_fp8_k<<<(128 * 2048) / 256, 256, 0, stream>>>(wf, thr + 4, twfB, 128, 2048, 2048);
    tern_f32_k<<<1, 256, 0, stream>>>(bf, thr + 4, tbf, 128);

    conv1_k<<<(4096 * 32 * 6 * 64) / 256, 256, 0, stream>>>(x, tw1, tb1, bufA);
    stats_init_k<<<1, 128, 0, stream>>>(sums, sqs);
    stats_k<<<dim3(256, 32), 256, 0, stream>>>(bufA, 4096, 32, 384, sums, sqs);
    bn1_pool_tern_k<<<(4096 * 32 * 6 * 32) / 256, 256, 0, stream>>>(
        bufA, g1, be1, sums, sqs, thr + 1, act);

    im2col2_k<<<(786432 * 128) / 256, 256, 0, stream>>>(act, im);
    gemm_fp8_k<4><<<6144, 256, 0, stream>>>(im, tw2B, tb2, bufA, 786432, 128);
    stats_init_k<<<1, 128, 0, stream>>>(sums, sqs);
    stats_k<<<dim3(256, 64), 256, 0, stream>>>(bufA, 786432, 64, 1, sums, sqs);
    bn2_tern_k<<<(4096 * 64 * 6 * 32) / 256, 256, 0, stream>>>(
        bufA, g2, be2, sums, sqs, thr + 2, act);

    im2col3_k<<<(786432 * 256) / 256, 256, 0, stream>>>(act, im);
    gemm_fp8_k<8><<<6144, 256, 0, stream>>>(im, tw3B, tb3, bufA, 786432, 256);
    stats_init_k<<<1, 128, 0, stream>>>(sums, sqs);
    stats_k<<<dim3(256, 128), 256, 0, stream>>>(bufA, 786432, 128, 1, sums, sqs);
    bn3_pool_tern_k<<<(4096 * 128 * 6 * 16) / 256, 256, 0, stream>>>(
        bufA, g3, be3, sums, sqs, thr + 3, act);

    im2col4_k<<<(65536 * 768) / 256, 256, 0, stream>>>(act, im);
    gemm_fp8_k<8><<<512, 256, 0, stream>>>(im, tw4B, tb4, bufA, 65536, 768);
    stats_init_k<<<1, 128, 0, stream>>>(sums, sqs);
    stats_k<<<dim3(256, 128), 256, 0, stream>>>(bufA, 65536, 128, 1, sums, sqs);
    bn4_tern_k<<<(4096 * 128 * 16) / 256, 256, 0, stream>>>(
        bufA, g4, be4, sums, sqs, thr + 4, act);

    gemm_fp8_k<8><<<32, 256, 0, stream>>>(act, twfB, tbf, out, 4096, 2048);
}
